// LocalMHA2d_21328807592345
// MI455X (gfx1250) — matrix operand order repaired and checked by reading, not compiled, not run on hardware
//
#include <hip/hip_runtime.h>
#include <hip/hip_bf16.h>

// ---------------------------------------------------------------------------
// Swin-style windowed MHA block for MI455X (gfx1250, wave32, WMMA).
// Pipeline: [cvt weights bf16] -> [LayerNorm->bf16] -> [QKV GEMM (WMMA bf16)]
//           -> [windowed attention (WMMA bf16 + register softmax + LDS
//              C->A relayout)] -> [out GEMM + residual (WMMA bf16, f32 out)].
// ---------------------------------------------------------------------------

#define DIMC   256
#define DHD    64
#define NHEADS 4
#define IMG    256            // H == W == 256
#define NB     4
#define NPIX   (NB * IMG * IMG)   // 262144 pixels

typedef __attribute__((ext_vector_type(16))) __bf16 bfrag;   // 16x16x32 A/B frag
typedef __attribute__((ext_vector_type(8)))  __bf16 bhalf8;  // 16-byte chunk
typedef __attribute__((ext_vector_type(8)))  float  v8f;     // 16x16 C/D frag

__device__ __forceinline__ unsigned short f32_to_bf16(float f) {
  union { float f; unsigned u; } v; v.f = f;
  unsigned r = v.u + 0x7FFFu + ((v.u >> 16) & 1u);   // round-to-nearest-even
  return (unsigned short)(r >> 16);
}

// A fragment (16x32, bf16, row-major source, leading dim `ld` elements).
// Lane layout: lanes 0-15 -> rows 0-15 with K {0..7,16..23};
//              lanes 16-31 -> rows 0-15 with K {8..15,24..31}.
__device__ __forceinline__ bfrag load_a_frag(const unsigned short* p, int ld) {
  const int lane = threadIdx.x & 31;
  const int m  = lane & 15;
  const int kb = (lane >> 4) << 3;                // 0 or 8
  const unsigned short* q = p + (size_t)m * ld + kb;
  union { bfrag v; bhalf8 h[2]; } u;
  u.h[0] = *(const bhalf8*)(const void*)(q);       // K = kb..kb+7
  u.h[1] = *(const bhalf8*)(const void*)(q + 16);  // K = kb+16..kb+23
  return u.v;
}

// B fragment (32x16) where B[k][n] = Src[n][k], Src row-major with leading
// dim `ld`.  Lane layout: lanes 0-15 -> n, K {0..7,16..23}; lanes 16-31 -> n, K {8..15,24..31}.
__device__ __forceinline__ bfrag load_b_frag(const unsigned short* p, int ld) {
  const int lane = threadIdx.x & 31;
  const int n   = lane & 15;
  const int klo = (lane >> 4) << 3;               // 0 or 8
  const unsigned short* q = p + (size_t)n * ld + klo;
  union { bfrag v; bhalf8 h[2]; } u;
  u.h[0] = *(const bhalf8*)(const void*)(q);
  u.h[1] = *(const bhalf8*)(const void*)(q + 16);
  return u.v;
}

__device__ __forceinline__ v8f wmma_bf16(bfrag a, bfrag b, v8f c) {
  return __builtin_amdgcn_wmma_f32_16x16x32_bf16(
      false, a, false, b, (short)0, c, false, false);
}

// ---------------------------------------------------------------------------
// Kernel 0: convert Wqkv (768x256) and Wout (256x256) to bf16.
// ---------------------------------------------------------------------------
__global__ __launch_bounds__(256)
void cvt_weights_kernel(const float* __restrict__ wqkv,
                        const float* __restrict__ wout,
                        unsigned short* __restrict__ wq_bf,
                        unsigned short* __restrict__ wo_bf) {
  const int i = blockIdx.x * 256 + threadIdx.x;   // 0..262143
  if (i < 3 * DIMC * DIMC) {
    wq_bf[i] = f32_to_bf16(wqkv[i]);
  } else {
    const int j = i - 3 * DIMC * DIMC;            // 0..65535
    wo_bf[j] = f32_to_bf16(wout[j]);
  }
}

// ---------------------------------------------------------------------------
// Kernel 1: LayerNorm over channels (NCHW input), emit bf16 xn[pixel][c].
// ---------------------------------------------------------------------------
__global__ __launch_bounds__(256)
void layernorm_kernel(const float* __restrict__ x,
                      const float* __restrict__ gamma,
                      const float* __restrict__ beta,
                      unsigned short* __restrict__ xn) {
  const int p  = blockIdx.x * 256 + threadIdx.x;  // pixel id
  const int b  = p >> 16;
  const int hw = p & 65535;
  const float* xp = x + (size_t)b * DIMC * (IMG * IMG) + hw;

  float s = 0.f, ss = 0.f;
  #pragma unroll 8
  for (int c = 0; c < DIMC; ++c) {
    float v = xp[(size_t)c * (IMG * IMG)];
    s += v; ss += v * v;
  }
  const float mean = s * (1.f / DIMC);
  const float var  = ss * (1.f / DIMC) - mean * mean;
  const float rstd = rsqrtf(var + 1e-5f);

  unsigned short* o = xn + (size_t)p * DIMC;
  #pragma unroll 8
  for (int c = 0; c < DIMC; ++c) {
    float v = xp[(size_t)c * (IMG * IMG)];
    o[c] = f32_to_bf16((v - mean) * rstd * gamma[c] + beta[c]);
  }
}

// ---------------------------------------------------------------------------
// Kernel 2: QKV GEMM.  (262144 x 256) x (256 x 768), bf16 in, f32 acc.
// Block = 8 waves arranged 2(M) x 4(N); wave tile = 32x32 (2x2 WMMA frags).
// Epilogue scatters into window-major q (pre-scaled), k, and transposed v.
// ---------------------------------------------------------------------------
__global__ __launch_bounds__(256)
void qkv_gemm_kernel(const unsigned short* __restrict__ xn,
                     const unsigned short* __restrict__ wq,
                     unsigned short* __restrict__ qb,
                     unsigned short* __restrict__ kb,
                     unsigned short* __restrict__ vtb) {
  const int lane  = threadIdx.x & 31;
  const int wave  = threadIdx.x >> 5;
  const int row0  = blockIdx.x * 64 + (wave >> 2) * 32;   // pixel row base
  const int col0  = blockIdx.y * 128 + (wave & 3) * 32;   // feature base

  v8f acc[2][2];
  #pragma unroll
  for (int i = 0; i < 2; ++i)
    #pragma unroll
    for (int j = 0; j < 2; ++j) acc[i][j] = {};

  #pragma unroll
  for (int kt = 0; kt < 8; ++kt) {                // K = 256, 32 per step
    bfrag a0 = load_a_frag(xn + (size_t)row0 * DIMC + kt * 32, DIMC);
    bfrag a1 = load_a_frag(xn + (size_t)(row0 + 16) * DIMC + kt * 32, DIMC);
    bfrag b0 = load_b_frag(wq + (size_t)col0 * DIMC + kt * 32, DIMC);
    bfrag b1 = load_b_frag(wq + (size_t)(col0 + 16) * DIMC + kt * 32, DIMC);
    acc[0][0] = wmma_bf16(a0, b0, acc[0][0]);
    acc[0][1] = wmma_bf16(a0, b1, acc[0][1]);
    acc[1][0] = wmma_bf16(a1, b0, acc[1][0]);
    acc[1][1] = wmma_bf16(a1, b1, acc[1][1]);
  }

  const int nl = lane & 15;
  const int lg = lane >> 4;
  #pragma unroll
  for (int mt = 0; mt < 2; ++mt) {
    #pragma unroll
    for (int nt = 0; nt < 2; ++nt) {
      #pragma unroll
      for (int j = 0; j < 8; ++j) {
        const int row = row0 + mt * 16 + j + 8 * lg;       // pixel
        const int f   = col0 + nt * 16 + nl;               // feature 0..767
        const int bi  = row >> 16;
        const int hw  = row & 65535;
        const int h   = hw >> 8, w = hw & 255;
        const int r   = ((h & 7) << 3) | (w & 7);          // row in window
        const int wy  = h >> 3, wx = w >> 3;
        const int kind = f >> 8;                           // 0=q 1=k 2=v
        const int c    = f & 255;
        const int head = c >> 6, d = c & 63;
        const size_t wb =
            ((size_t)(((bi * 32 + wy) * 32 + wx) * NHEADS + head)) * 4096;
        const float v = acc[mt][nt][j];
        if (kind == 0)      qb[wb + r * 64 + d] = f32_to_bf16(v * 0.125f);
        else if (kind == 1) kb[wb + r * 64 + d] = f32_to_bf16(v);
        else                vtb[wb + d * 64 + r] = f32_to_bf16(v);  // V^T
      }
    }
  }
}

// ---------------------------------------------------------------------------
// Kernel 3: windowed attention.  One wave per (window, head).
// S = Q K^T (4x4x2 WMMA), register softmax via half-wave shuffles,
// P staged through LDS (C-layout -> A-layout), O = P V (WMMA vs V^T).
// ---------------------------------------------------------------------------
__global__ __launch_bounds__(128)
void attn_kernel(const unsigned short* __restrict__ qb,
                 const unsigned short* __restrict__ kb,
                 const unsigned short* __restrict__ vtb,
                 unsigned short* __restrict__ ob) {
  __shared__ unsigned short Plds[4 * 4096];       // 8 KB per wave, 32 KB total

  const int lane = threadIdx.x & 31;
  const int wave = threadIdx.x >> 5;
  const int wh   = blockIdx.x * 4 + wave;         // 0..16383
  const int head = wh & 3;
  const int win  = wh >> 2;
  const int wx   = win & 31;
  const int wy   = (win >> 5) & 31;
  const int bi   = win >> 10;
  const size_t base = (size_t)wh * 4096;          // 64x64 bf16 tiles

  // ---- S = Q * K^T -------------------------------------------------------
  v8f s[4][4];
  #pragma unroll
  for (int i = 0; i < 4; ++i)
    #pragma unroll
    for (int j = 0; j < 4; ++j) s[i][j] = {};

  #pragma unroll
  for (int kt = 0; kt < 2; ++kt) {
    bfrag aq[4];
    #pragma unroll
    for (int mt = 0; mt < 4; ++mt)
      aq[mt] = load_a_frag(qb + base + (mt * 16) * 64 + kt * 32, 64);
    #pragma unroll
    for (int nt = 0; nt < 4; ++nt) {
      bfrag bk = load_b_frag(kb + base + (nt * 16) * 64 + kt * 32, 64);
      #pragma unroll
      for (int mt = 0; mt < 4; ++mt)
        s[mt][nt] = wmma_bf16(aq[mt], bk, s[mt][nt]);
    }
  }

  // ---- softmax over each 64-wide row (rows live in 16-lane halves) -------
  const int nl = lane & 15;
  const int lg = lane >> 4;
  unsigned short* pw = &Plds[wave * 4096];
  #pragma unroll
  for (int mt = 0; mt < 4; ++mt) {
    #pragma unroll
    for (int j = 0; j < 8; ++j) {
      float m0 = s[mt][0][j];
      #pragma unroll
      for (int nt = 1; nt < 4; ++nt) m0 = fmaxf(m0, s[mt][nt][j]);
      #pragma unroll
      for (int msk = 1; msk < 16; msk <<= 1)
        m0 = fmaxf(m0, __shfl_xor(m0, msk, 32));
      float e[4], sum = 0.f;
      #pragma unroll
      for (int nt = 0; nt < 4; ++nt) { e[nt] = __expf(s[mt][nt][j] - m0); sum += e[nt]; }
      #pragma unroll
      for (int msk = 1; msk < 16; msk <<= 1)
        sum += __shfl_xor(sum, msk, 32);
      const float inv = __frcp_rn(sum);
      const int row = mt * 16 + j + 8 * lg;
      #pragma unroll
      for (int nt = 0; nt < 4; ++nt)
        pw[row * 64 + nt * 16 + nl] = f32_to_bf16(e[nt] * inv);   // ds_store
    }
  }
  __syncthreads();

  // ---- O = P * V  (A from LDS, B from contiguous V^T) --------------------
  v8f o[4][4];
  #pragma unroll
  for (int i = 0; i < 4; ++i)
    #pragma unroll
    for (int j = 0; j < 4; ++j) o[i][j] = {};

  const int am = lane & 15;
  const int akb = (lane >> 4) << 3;
  #pragma unroll
  for (int kt = 0; kt < 2; ++kt) {
    bfrag ap[4];
    #pragma unroll
    for (int mt = 0; mt < 4; ++mt) {
      const unsigned short* q = &pw[(mt * 16 + am) * 64 + kt * 32 + akb];
      union { bfrag v; bhalf8 h[2]; } u;
      u.h[0] = *(const bhalf8*)(const void*)(q);        // ds_load_b128
      u.h[1] = *(const bhalf8*)(const void*)(q + 16);
      ap[mt] = u.v;
    }
    #pragma unroll
    for (int nt = 0; nt < 4; ++nt) {
      bfrag bv = load_b_frag(vtb + base + (nt * 16) * 64 + kt * 32, 64);
      #pragma unroll
      for (int mt = 0; mt < 4; ++mt)
        o[mt][nt] = wmma_bf16(ap[mt], bv, o[mt][nt]);
    }
  }

  // ---- write O back to pixel-major bf16 [pixel][c] -----------------------
  #pragma unroll
  for (int mt = 0; mt < 4; ++mt) {
    #pragma unroll
    for (int nt = 0; nt < 4; ++nt) {
      #pragma unroll
      for (int j = 0; j < 8; ++j) {
        const int r = mt * 16 + j + 8 * lg;            // row in window
        const int h = wy * 8 + (r >> 3);
        const int w = wx * 8 + (r & 7);
        const size_t p = ((size_t)bi * IMG + h) * IMG + w;
        const int c = head * 64 + nt * 16 + nl;
        ob[p * DIMC + c] = f32_to_bf16(o[mt][nt][j]);
      }
    }
  }
}

// ---------------------------------------------------------------------------
// Kernel 4: output projection (262144x256 x 256x256) + residual, f32 NCHW out.
// ---------------------------------------------------------------------------
__global__ __launch_bounds__(256)
void out_gemm_kernel(const unsigned short* __restrict__ ob,
                     const unsigned short* __restrict__ wo,
                     const float* __restrict__ x,
                     float* __restrict__ out) {
  const int lane = threadIdx.x & 31;
  const int wave = threadIdx.x >> 5;
  const int row0 = blockIdx.x * 64 + (wave >> 2) * 32;
  const int col0 = blockIdx.y * 128 + (wave & 3) * 32;

  v8f acc[2][2];
  #pragma unroll
  for (int i = 0; i < 2; ++i)
    #pragma unroll
    for (int j = 0; j < 2; ++j) acc[i][j] = {};

  #pragma unroll
  for (int kt = 0; kt < 8; ++kt) {
    bfrag a0 = load_a_frag(ob + (size_t)row0 * DIMC + kt * 32, DIMC);
    bfrag a1 = load_a_frag(ob + (size_t)(row0 + 16) * DIMC + kt * 32, DIMC);
    bfrag b0 = load_b_frag(wo + (size_t)col0 * DIMC + kt * 32, DIMC);
    bfrag b1 = load_b_frag(wo + (size_t)(col0 + 16) * DIMC + kt * 32, DIMC);
    acc[0][0] = wmma_bf16(a0, b0, acc[0][0]);
    acc[0][1] = wmma_bf16(a0, b1, acc[0][1]);
    acc[1][0] = wmma_bf16(a1, b0, acc[1][0]);
    acc[1][1] = wmma_bf16(a1, b1, acc[1][1]);
  }

  const int nl = lane & 15;
  const int lg = lane >> 4;
  #pragma unroll
  for (int mt = 0; mt < 2; ++mt) {
    #pragma unroll
    for (int nt = 0; nt < 2; ++nt) {
      #pragma unroll
      for (int j = 0; j < 8; ++j) {
        const int row = row0 + mt * 16 + j + 8 * lg;   // pixel
        const int f   = col0 + nt * 16 + nl;           // out channel
        const int bi  = row >> 16;
        const int hw  = row & 65535;
        const size_t oidx = ((size_t)(bi * DIMC + f)) * (IMG * IMG) + hw;
        out[oidx] = acc[mt][nt][j] + x[oidx];          // + residual
      }
    }
  }
}

// ---------------------------------------------------------------------------
// Host-side launcher.
// ---------------------------------------------------------------------------
extern "C" void kernel_launch(void* const* d_in, const int* in_sizes, int n_in,
                              void* d_out, int out_size, void* d_ws, size_t ws_size,
                              hipStream_t stream) {
  const float* x     = (const float*)d_in[0];
  const float* gamma = (const float*)d_in[1];
  const float* beta  = (const float*)d_in[2];
  const float* Wqkv  = (const float*)d_in[3];
  const float* Wout  = (const float*)d_in[4];
  float* out = (float*)d_out;

  // Workspace layout (bytes).  xn buffer is reused as attention output `ob`
  // (xn's last consumer is the QKV GEMM, which completes before attention).
  char* ws = (char*)d_ws;
  const size_t ACT = (size_t)NPIX * DIMC * sizeof(unsigned short); // 128 MB
  unsigned short* wq_bf = (unsigned short*)(ws);                   // 384 KB
  unsigned short* wo_bf = (unsigned short*)(ws + 393216);          // 128 KB
  unsigned short* xn    = (unsigned short*)(ws + 524288);
  unsigned short* ob    = xn;                                      // aliased
  unsigned short* qb    = (unsigned short*)(ws + 524288 + ACT);
  unsigned short* kb    = (unsigned short*)(ws + 524288 + 2 * ACT);
  unsigned short* vtb   = (unsigned short*)(ws + 524288 + 3 * ACT);

  cvt_weights_kernel<<<dim3(1024), dim3(256), 0, stream>>>(Wqkv, Wout, wq_bf, wo_bf);
  layernorm_kernel<<<dim3(NPIX / 256), dim3(256), 0, stream>>>(x, gamma, beta, xn);
  qkv_gemm_kernel<<<dim3(NPIX / 64, 6), dim3(256), 0, stream>>>(xn, wq_bf, qb, kb, vtb);
  attn_kernel<<<dim3(NB * 1024 * NHEADS / 4), dim3(128), 0, stream>>>(qb, kb, vtb, ob);
  out_gemm_kernel<<<dim3(NPIX / 64, 2), dim3(256), 0, stream>>>(ob, wo_bf, x, out);
}
